// SACCrystalActor_71829033058348
// MI455X (gfx1250) — matrix operand order repaired and checked by reading, not compiled, not run on hardware
//
#include <hip/hip_runtime.h>

typedef __bf16 v16bf __attribute__((ext_vector_type(16)));
typedef float  v8f   __attribute__((ext_vector_type(8)));

#define H    128
#define TM   16
#define TMB  64
#define APAD 8

__device__ inline unsigned short f32_to_bf16(float f) {
  unsigned int u = __float_as_uint(f);
  unsigned int r = (u + 0x7FFFu + ((u >> 16) & 1u)) >> 16;
  return (unsigned short)r;
}

__device__ inline unsigned int f32_ord(float f) {
  unsigned int u = __float_as_uint(f);
  return (u & 0x80000000u) ? ~u : (u | 0x80000000u);
}
__device__ inline float ord_f32(unsigned int u) {
  unsigned int v = (u & 0x80000000u) ? (u & 0x7FFFFFFFu) : ~u;
  return __uint_as_float(v);
}

union BfRegs { v16bf v; unsigned int u[8]; };

// A operand, contiguous k order used here (B packed to match): lane L -> M = L%16, kbase = (L/16)*16;
// V0..3 hold K = kbase..kbase+7 (2 per VGPR), V4..7 hold K = kbase+8..kbase+15.
__device__ inline v16bf load_a_lds(const unsigned short* row, int kc, int kh) {
  BfRegs r;
  const unsigned int* p0 = (const unsigned int*)(row + kc * 32 + kh * 16);
  const unsigned int* p1 = (const unsigned int*)(row + kc * 32 + kh * 16 + 8);
#pragma unroll
  for (int i = 0; i < 4; ++i) { r.u[i] = p0[i]; r.u[4 + i] = p1[i]; }
  return r.v;
}

// B tiles pre-packed as the per-lane register image: tile*32 lanes * 8 dwords.
__device__ inline v16bf load_b_packed(const unsigned int* tiles, int tile, int lane) {
  BfRegs r;
  const unsigned int* p = tiles + ((size_t)tile * 32 + lane) * 8;
#pragma unroll
  for (int i = 0; i < 8; ++i) r.u[i] = p[i];
  return r.v;
}

__device__ inline v8f wmma_bf16(v16bf a, v16bf b, v8f c) {
  return __builtin_amdgcn_wmma_f32_16x16x32_bf16(false, a, false, b, (short)0, c,
                                                 false, false);
}

// Pack row-major W[K][128] (f32) into WMMA B-operand tiles (bf16).
// grid.x = (K/32)*8 tiles, block = 32.
// B k order used here (contiguous, as load_a_lds): lane L -> N = L%16; L<16: K=0..15 in V0..7 (2/VGPR);
// L>=16: K=16..31 in V0..7.
__global__ void pack_b_kernel(const float* __restrict__ w, unsigned int* __restrict__ out) {
  const int tile = blockIdx.x;
  const int kc = tile >> 3;
  const int nt = tile & 7;
  const int lane = threadIdx.x;
  const int n = nt * 16 + (lane & 15);
  const int kbase = kc * 32 + (lane >> 4) * 16;
  unsigned int* o = out + ((size_t)tile * 32 + lane) * 8;
#pragma unroll
  for (int v = 0; v < 8; ++v) {
    const int k = kbase + 2 * v;
    unsigned int lo = f32_to_bf16(w[(size_t)k * H + n]);
    unsigned int hi = f32_to_bf16(w[(size_t)(k + 1) * H + n]);
    o[v] = lo | (hi << 16);
  }
}

// ---------------- Node encoder: h = relu(nf@W1+b1)@W2+b2 -------------------
__global__ __launch_bounds__(128) void nodeenc_kernel(
    const float* __restrict__ nf, const float* __restrict__ w1,
    const float* __restrict__ b1, const unsigned int* __restrict__ w2p,
    const float* __restrict__ b2, float* __restrict__ h, int N_) {
  __shared__ __align__(16) unsigned short Ab[TM][H + APAD];
  const int tid = threadIdx.x, lane = tid & 31, wave = tid >> 5;
  const int m = lane & 15, kh = lane >> 4;
  const int n0 = blockIdx.x * TM;
  {
    const int row = tid >> 3, sub = tid & 7;
    const int node = n0 + row;
    const bool v = node < N_;
#pragma unroll
    for (int j = 0; j < 16; ++j) {
      const int c = sub * 16 + j;
      float s = b1[c];
      if (v) {
#pragma unroll
        for (int k = 0; k < 7; ++k) s += nf[(size_t)node * 7 + k] * w1[k * H + c];
      }
      s = s > 0.f ? s : 0.f;
      Ab[row][c] = f32_to_bf16(s);
    }
  }
  __syncthreads();
  const v8f vz = {0, 0, 0, 0, 0, 0, 0, 0};
  v8f a0 = vz, a1 = vz;
#pragma unroll
  for (int kc = 0; kc < 4; ++kc) {
    v16bf a = load_a_lds(&Ab[m][0], kc, kh);
    a0 = wmma_bf16(a, load_b_packed(w2p, kc * 8 + wave * 2, lane), a0);
    a1 = wmma_bf16(a, load_b_packed(w2p, kc * 8 + wave * 2 + 1, lane), a1);
  }
  const int c0 = wave * 32 + m;
  const float bi0 = b2[c0], bi1 = b2[c0 + 16];
#pragma unroll
  for (int r = 0; r < 8; ++r) {
    const int node = n0 + r + 8 * kh;
    if (node < N_) {
      h[(size_t)node * H + c0] = a0[r] + bi0;
      h[(size_t)node * H + c0 + 16] = a1[r] + bi1;
    }
  }
}

// ---------------- Edge encoder: e = relu(ef@W1+b1)@W2+b2 (bf16 out) --------
__global__ __launch_bounds__(128) void edgeenc_kernel(
    const float* __restrict__ ef, const float* __restrict__ w1,
    const float* __restrict__ b1, const unsigned int* __restrict__ w2p,
    const float* __restrict__ b2, unsigned short* __restrict__ e, int E_) {
  __shared__ __align__(16) unsigned short Ab[TM][H + APAD];
  const int tid = threadIdx.x, lane = tid & 31, wave = tid >> 5;
  const int m = lane & 15, kh = lane >> 4;
  const int e0 = blockIdx.x * TM;
  {
    const int row = tid >> 3, sub = tid & 7;
    const int ei = e0 + row;
    const bool v = ei < E_;
    const float f0 = v ? ef[(size_t)ei * 2] : 0.f;
    const float f1 = v ? ef[(size_t)ei * 2 + 1] : 0.f;
#pragma unroll
    for (int j = 0; j < 16; ++j) {
      const int c = sub * 16 + j;
      float s = b1[c] + f0 * w1[c] + f1 * w1[H + c];
      s = s > 0.f ? s : 0.f;
      Ab[row][c] = f32_to_bf16(s);
    }
  }
  __syncthreads();
  const v8f vz = {0, 0, 0, 0, 0, 0, 0, 0};
  v8f a0 = vz, a1 = vz;
#pragma unroll
  for (int kc = 0; kc < 4; ++kc) {
    v16bf a = load_a_lds(&Ab[m][0], kc, kh);
    a0 = wmma_bf16(a, load_b_packed(w2p, kc * 8 + wave * 2, lane), a0);
    a1 = wmma_bf16(a, load_b_packed(w2p, kc * 8 + wave * 2 + 1, lane), a1);
  }
  const int c0 = wave * 32 + m;
  const float bi0 = b2[c0], bi1 = b2[c0 + 16];
#pragma unroll
  for (int r = 0; r < 8; ++r) {
    const int ei = e0 + r + 8 * kh;
    if (ei < E_) {
      e[(size_t)ei * H + c0] = f32_to_bf16(a0[r] + bi0);
      e[(size_t)ei * H + c0 + 16] = f32_to_bf16(a1[r] + bi1);
    }
  }
}

// ------------- Message MLP + scatter-add: m = MLP2([h_s|h_d|e]) ------------
// 64 edges/block (4 row-subtiles of 16) to amortize packed-weight loads 4x.
// Layer-2 activation buffer overlays the (dead) layer-1 A buffer to stay
// under the static LDS limit.
__global__ __launch_bounds__(128) void msg_kernel(
    const float* __restrict__ h, const unsigned short* __restrict__ e,
    const int* __restrict__ src, const int* __restrict__ dst,
    const unsigned int* __restrict__ w1p, const float* __restrict__ b1,
    const unsigned int* __restrict__ w2p, const float* __restrict__ b2,
    float* __restrict__ hagg, int E_) {
  __shared__ __align__(16) unsigned short Ab[TMB][3 * H + APAD];  // 50176 B
  __shared__ int dIds[TMB];
  unsigned short (*Hb)[H + APAD] = (unsigned short (*)[H + APAD]) & Ab[0][0];

  const int tid = threadIdx.x, lane = tid & 31, wave = tid >> 5;
  const int m = lane & 15, kh = lane >> 4;
  const int e0 = blockIdx.x * TMB;

#pragma unroll
  for (int g = 0; g < 4; ++g) {
    const int row = g * 16 + (tid >> 3), sub = tid & 7;
    const int ei = e0 + row;
    const bool v = ei < E_;
    const int s = v ? src[ei] : 0;
    const int d = v ? dst[ei] : 0;
    if (sub == 0) dIds[row] = d;
    const float* hs = h + (size_t)s * H;
    const float* hd = h + (size_t)d * H;
    const unsigned short* ep = e + (size_t)ei * H;
#pragma unroll
    for (int j = 0; j < 16; ++j) {
      const int c = sub * 16 + j;
      Ab[row][c]         = v ? f32_to_bf16(hs[c]) : (unsigned short)0;
      Ab[row][H + c]     = v ? f32_to_bf16(hd[c]) : (unsigned short)0;
      Ab[row][2 * H + c] = v ? ep[c] : (unsigned short)0;
    }
  }
  __syncthreads();

  const v8f vz = {0, 0, 0, 0, 0, 0, 0, 0};
  v8f acc[4][2];
#pragma unroll
  for (int rt = 0; rt < 4; ++rt) { acc[rt][0] = vz; acc[rt][1] = vz; }
#pragma unroll
  for (int kc = 0; kc < 12; ++kc) {
    v16bf b0 = load_b_packed(w1p, kc * 8 + wave * 2, lane);
    v16bf bb = load_b_packed(w1p, kc * 8 + wave * 2 + 1, lane);
#pragma unroll
    for (int rt = 0; rt < 4; ++rt) {
      v16bf a = load_a_lds(&Ab[rt * 16 + m][0], kc, kh);
      acc[rt][0] = wmma_bf16(a, b0, acc[rt][0]);
      acc[rt][1] = wmma_bf16(a, bb, acc[rt][1]);
    }
  }
  __syncthreads();  // all reads of Ab done before Hb overlay-writes

  const int c0 = wave * 32 + m;
  {
    const float bi0 = b1[c0], bi1 = b1[c0 + 16];
#pragma unroll
    for (int rt = 0; rt < 4; ++rt) {
#pragma unroll
      for (int r = 0; r < 8; ++r) {
        const int row = rt * 16 + r + 8 * kh;
        float x0 = acc[rt][0][r] + bi0; x0 = x0 > 0.f ? x0 : 0.f;
        float x1 = acc[rt][1][r] + bi1; x1 = x1 > 0.f ? x1 : 0.f;
        Hb[row][c0] = f32_to_bf16(x0);
        Hb[row][c0 + 16] = f32_to_bf16(x1);
      }
    }
  }
  __syncthreads();

  v8f d2[4][2];
#pragma unroll
  for (int rt = 0; rt < 4; ++rt) { d2[rt][0] = vz; d2[rt][1] = vz; }
#pragma unroll
  for (int kc = 0; kc < 4; ++kc) {
    v16bf b0 = load_b_packed(w2p, kc * 8 + wave * 2, lane);
    v16bf bb = load_b_packed(w2p, kc * 8 + wave * 2 + 1, lane);
#pragma unroll
    for (int rt = 0; rt < 4; ++rt) {
      v16bf a = load_a_lds(&Hb[rt * 16 + m][0], kc, kh);
      d2[rt][0] = wmma_bf16(a, b0, d2[rt][0]);
      d2[rt][1] = wmma_bf16(a, bb, d2[rt][1]);
    }
  }
  {
    const float bi0 = b2[c0], bi1 = b2[c0 + 16];
#pragma unroll
    for (int rt = 0; rt < 4; ++rt) {
#pragma unroll
      for (int r = 0; r < 8; ++r) {
        const int row = rt * 16 + r + 8 * kh;
        const int ei = e0 + row;
        if (ei < E_) {
          const int d = dIds[row];
          atomicAdd(&hagg[(size_t)d * H + c0], d2[rt][0][r] + bi0);
          atomicAdd(&hagg[(size_t)d * H + c0 + 16], d2[rt][1][r] + bi1);
        }
      }
    }
  }
}

// ------------- Update MLP + residual: h = MLP2([h|h_agg]) + h --------------
__global__ __launch_bounds__(128) void upd_kernel(
    float* __restrict__ h, const float* __restrict__ hagg,
    const unsigned int* __restrict__ w1p, const float* __restrict__ b1,
    const unsigned int* __restrict__ w2p, const float* __restrict__ b2, int N_) {
  __shared__ __align__(16) unsigned short Ab[TMB][2 * H + APAD];  // 33792 B
  unsigned short (*Hb)[H + APAD] = (unsigned short (*)[H + APAD]) & Ab[0][0];
  const int tid = threadIdx.x, lane = tid & 31, wave = tid >> 5;
  const int m = lane & 15, kh = lane >> 4;
  const int n0 = blockIdx.x * TMB;

#pragma unroll
  for (int g = 0; g < 4; ++g) {
    const int row = g * 16 + (tid >> 3), sub = tid & 7;
    const int node = n0 + row;
    const bool v = node < N_;
    const float* hp = h + (size_t)(v ? node : 0) * H;
    const float* gp = hagg + (size_t)(v ? node : 0) * H;
#pragma unroll
    for (int j = 0; j < 16; ++j) {
      const int c = sub * 16 + j;
      Ab[row][c]     = v ? f32_to_bf16(hp[c]) : (unsigned short)0;
      Ab[row][H + c] = v ? f32_to_bf16(gp[c]) : (unsigned short)0;
    }
  }
  __syncthreads();

  const v8f vz = {0, 0, 0, 0, 0, 0, 0, 0};
  v8f acc[4][2];
#pragma unroll
  for (int rt = 0; rt < 4; ++rt) { acc[rt][0] = vz; acc[rt][1] = vz; }
#pragma unroll
  for (int kc = 0; kc < 8; ++kc) {
    v16bf b0 = load_b_packed(w1p, kc * 8 + wave * 2, lane);
    v16bf bb = load_b_packed(w1p, kc * 8 + wave * 2 + 1, lane);
#pragma unroll
    for (int rt = 0; rt < 4; ++rt) {
      v16bf a = load_a_lds(&Ab[rt * 16 + m][0], kc, kh);
      acc[rt][0] = wmma_bf16(a, b0, acc[rt][0]);
      acc[rt][1] = wmma_bf16(a, bb, acc[rt][1]);
    }
  }
  __syncthreads();  // Ab reads done before overlay

  const int c0 = wave * 32 + m;
  {
    const float bi0 = b1[c0], bi1 = b1[c0 + 16];
#pragma unroll
    for (int rt = 0; rt < 4; ++rt) {
#pragma unroll
      for (int r = 0; r < 8; ++r) {
        const int row = rt * 16 + r + 8 * kh;
        float x0 = acc[rt][0][r] + bi0; x0 = x0 > 0.f ? x0 : 0.f;
        float x1 = acc[rt][1][r] + bi1; x1 = x1 > 0.f ? x1 : 0.f;
        Hb[row][c0] = f32_to_bf16(x0);
        Hb[row][c0 + 16] = f32_to_bf16(x1);
      }
    }
  }
  __syncthreads();

  v8f d2[4][2];
#pragma unroll
  for (int rt = 0; rt < 4; ++rt) { d2[rt][0] = vz; d2[rt][1] = vz; }
#pragma unroll
  for (int kc = 0; kc < 4; ++kc) {
    v16bf b0 = load_b_packed(w2p, kc * 8 + wave * 2, lane);
    v16bf bb = load_b_packed(w2p, kc * 8 + wave * 2 + 1, lane);
#pragma unroll
    for (int rt = 0; rt < 4; ++rt) {
      v16bf a = load_a_lds(&Hb[rt * 16 + m][0], kc, kh);
      d2[rt][0] = wmma_bf16(a, b0, d2[rt][0]);
      d2[rt][1] = wmma_bf16(a, bb, d2[rt][1]);
    }
  }
  {
    const float bi0 = b2[c0], bi1 = b2[c0 + 16];
#pragma unroll
    for (int rt = 0; rt < 4; ++rt) {
#pragma unroll
      for (int r = 0; r < 8; ++r) {
        const int node = n0 + rt * 16 + r + 8 * kh;
        if (node < N_) {
          float* p0 = &h[(size_t)node * H + c0];
          float* p1 = &h[(size_t)node * H + c0 + 16];
          *p0 = d2[rt][0][r] + bi0 + *p0;
          *p1 = d2[rt][1][r] + bi1 + *p1;
        }
      }
    }
  }
}

// ------------- Gate: gate = relu(h@W1+b1)@w2 + b2 ; atomicMax per graph ----
__global__ __launch_bounds__(128) void gate_kernel(
    const float* __restrict__ h, const unsigned int* __restrict__ w1p,
    const float* __restrict__ b1, const float* __restrict__ w2,
    const float* __restrict__ b2, const int* __restrict__ gid,
    float* __restrict__ gate, unsigned int* __restrict__ gmax, int N_) {
  __shared__ __align__(16) unsigned short Ab[TM][H + APAD];
  __shared__ float Hf[TM][H + 2];
  __shared__ float part[TM][8];
  const int tid = threadIdx.x, lane = tid & 31, wave = tid >> 5;
  const int m = lane & 15, kh = lane >> 4;
  const int n0 = blockIdx.x * TM;
  {
    const int row = tid >> 3, sub = tid & 7;
    const int node = n0 + row;
    const bool v = node < N_;
    const float* hp = h + (size_t)(v ? node : 0) * H;
#pragma unroll
    for (int j = 0; j < 16; ++j) {
      const int c = sub * 16 + j;
      Ab[row][c] = v ? f32_to_bf16(hp[c]) : (unsigned short)0;
    }
  }
  __syncthreads();
  const v8f vz = {0, 0, 0, 0, 0, 0, 0, 0};
  v8f a0 = vz, a1 = vz;
#pragma unroll
  for (int kc = 0; kc < 4; ++kc) {
    v16bf a = load_a_lds(&Ab[m][0], kc, kh);
    a0 = wmma_bf16(a, load_b_packed(w1p, kc * 8 + wave * 2, lane), a0);
    a1 = wmma_bf16(a, load_b_packed(w1p, kc * 8 + wave * 2 + 1, lane), a1);
  }
  {
    const int c0 = wave * 32 + m;
    const float bi0 = b1[c0], bi1 = b1[c0 + 16];
#pragma unroll
    for (int r = 0; r < 8; ++r) {
      const int row = r + 8 * kh;
      float x0 = a0[r] + bi0; Hf[row][c0] = x0 > 0.f ? x0 : 0.f;
      float x1 = a1[r] + bi1; Hf[row][c0 + 16] = x1 > 0.f ? x1 : 0.f;
    }
  }
  __syncthreads();
  {
    const int row = tid >> 3, sub = tid & 7;
    float s = 0.f;
#pragma unroll
    for (int j = 0; j < 16; ++j) s += Hf[row][sub * 16 + j] * w2[sub * 16 + j];
    part[row][sub] = s;
  }
  __syncthreads();
  {
    const int row = tid >> 3, sub = tid & 7;
    const int node = n0 + row;
    if (sub == 0 && node < N_) {
      float t = b2[0];
#pragma unroll
      for (int k = 0; k < 8; ++k) t += part[row][k];
      gate[node] = t;
      atomicMax(&gmax[gid[node]], f32_ord(t));
    }
  }
}

__global__ void exp_kernel(const float* __restrict__ gate, const int* __restrict__ gid,
                           const unsigned int* __restrict__ gmax,
                           float* __restrict__ exv, float* __restrict__ z, int N_) {
  const int i = blockIdx.x * 256 + threadIdx.x;
  if (i < N_) {
    const int g = gid[i];
    const float ex = expf(gate[i] - ord_f32(gmax[g]));
    exv[i] = ex;
    atomicAdd(&z[g], ex);
  }
}

__global__ void emb_kernel(const float* __restrict__ h, const float* __restrict__ exv,
                           const float* __restrict__ z, const int* __restrict__ gid,
                           float* __restrict__ emb, int N_) {
  const size_t idx = (size_t)blockIdx.x * 256 + threadIdx.x;
  if (idx < (size_t)N_ * H) {
    const int i = (int)(idx >> 7);
    const int c = (int)(idx & 127);
    const int g = gid[i];
    const float alpha = exv[i] / z[g];
    atomicAdd(&emb[g * H + c], alpha * h[idx]);
  }
}

// ------------- Policy head: (64,128) -> mean/std -------------
__global__ __launch_bounds__(256) void policy_kernel(
    const float* __restrict__ emb, const float* __restrict__ w1,
    const float* __restrict__ b1, const float* __restrict__ w2,
    const float* __restrict__ b2, const float* __restrict__ w3,
    const float* __restrict__ b3, float* __restrict__ out) {
  __shared__ float S[64 * H];
  const int tid = threadIdx.x;
  float o[32];
#pragma unroll
  for (int j = 0; j < 32; ++j) S[tid * 32 + j] = emb[tid * 32 + j];
  __syncthreads();
  for (int j = 0; j < 32; ++j) {
    const int oI = tid * 32 + j, r = oI >> 7, c = oI & 127;
    float s = b1[c];
    for (int k = 0; k < H; ++k) s += S[r * H + k] * w1[k * H + c];
    o[j] = s > 0.f ? s : 0.f;
  }
  __syncthreads();
#pragma unroll
  for (int j = 0; j < 32; ++j) S[tid * 32 + j] = o[j];
  __syncthreads();
  for (int j = 0; j < 32; ++j) {
    const int oI = tid * 32 + j, r = oI >> 7, c = oI & 127;
    float s = b2[c];
    for (int k = 0; k < H; ++k) s += S[r * H + k] * w2[k * H + c];
    o[j] = s > 0.f ? s : 0.f;
  }
  __syncthreads();
#pragma unroll
  for (int j = 0; j < 32; ++j) S[tid * 32 + j] = o[j];
  __syncthreads();
  for (int j = 0; j < 32; ++j) {
    const int oI = tid * 32 + j, r = oI >> 7, c = oI & 127;
    float s = b3[c];
    for (int k = 0; k < H; ++k) s += S[r * H + k] * w3[k * H + c];
    if (c < 64) {
      out[r * 64 + c] = s;  // mean
    } else {
      float ls = s < -20.f ? -20.f : (s > 2.f ? 2.f : s);
      out[64 * 64 + r * 64 + (c - 64)] = expf(ls);  // std
    }
  }
}

extern "C" void kernel_launch(void* const* d_in, const int* in_sizes, int n_in,
                              void* d_out, int out_size, void* d_ws, size_t ws_size,
                              hipStream_t stream) {
  (void)n_in; (void)out_size; (void)ws_size;
  const float* node_feat = (const float*)d_in[0];
  const float* edge_feat = (const float*)d_in[1];
  const float* ne_w1 = (const float*)d_in[2];
  const float* ne_b1 = (const float*)d_in[3];
  const float* ne_w2 = (const float*)d_in[4];
  const float* ne_b2 = (const float*)d_in[5];
  const float* ee_w1 = (const float*)d_in[6];
  const float* ee_b1 = (const float*)d_in[7];
  const float* ee_w2 = (const float*)d_in[8];
  const float* ee_b2 = (const float*)d_in[9];
  const float* msg_w1 = (const float*)d_in[10];
  const float* msg_b1 = (const float*)d_in[11];
  const float* msg_w2 = (const float*)d_in[12];
  const float* msg_b2 = (const float*)d_in[13];
  const float* upd_w1 = (const float*)d_in[14];
  const float* upd_b1 = (const float*)d_in[15];
  const float* upd_w2 = (const float*)d_in[16];
  const float* upd_b2 = (const float*)d_in[17];
  const float* gate_w1 = (const float*)d_in[18];
  const float* gate_b1 = (const float*)d_in[19];
  const float* gate_w2 = (const float*)d_in[20];
  const float* gate_b2 = (const float*)d_in[21];
  const float* pol_w1 = (const float*)d_in[22];
  const float* pol_b1 = (const float*)d_in[23];
  const float* pol_w2 = (const float*)d_in[24];
  const float* pol_b2 = (const float*)d_in[25];
  const float* pol_w3 = (const float*)d_in[26];
  const float* pol_b3 = (const float*)d_in[27];
  const int* src = (const int*)d_in[28];
  const int* dst = (const int*)d_in[29];
  const int* gid = (const int*)d_in[30];
  const int N_ = in_sizes[30];
  const int E_ = in_sizes[28];
  const int B_ = 64;

  char* ws = (char*)d_ws;
  size_t off = 0;
  auto take = [&](size_t bytes) -> char* {
    char* p = ws + off;
    off += (bytes + 255) & ~(size_t)255;
    return p;
  };
  float* h = (float*)take((size_t)N_ * H * 4);
  float* hagg = (float*)take((size_t)N_ * H * 4);
  unsigned short* e = (unsigned short*)take((size_t)E_ * H * 2);
  float* gate = (float*)take((size_t)N_ * 4);
  float* exv = (float*)take((size_t)N_ * 4);
  unsigned int* gmax = (unsigned int*)take(B_ * 4);
  float* z = (float*)take(B_ * 4);
  float* emb = (float*)take((size_t)B_ * H * 4);
  // packed B-operand tiles (dwords): tile = 256 dwords; tiles(K) = (K/32)*8
  const size_t T128 = 32 * 256, T256 = 64 * 256, T384 = 96 * 256;
  unsigned int* p_ne_w2 = (unsigned int*)take(T128 * 4);
  unsigned int* p_ee_w2 = (unsigned int*)take(T128 * 4);
  unsigned int* p_gate_w1 = (unsigned int*)take(T128 * 4);
  unsigned int* p_msg_w1 = (unsigned int*)take(4 * T384 * 4);
  unsigned int* p_msg_w2 = (unsigned int*)take(4 * T128 * 4);
  unsigned int* p_upd_w1 = (unsigned int*)take(4 * T256 * 4);
  unsigned int* p_upd_w2 = (unsigned int*)take(4 * T128 * 4);

  pack_b_kernel<<<32, 32, 0, stream>>>(ne_w2, p_ne_w2);
  pack_b_kernel<<<32, 32, 0, stream>>>(ee_w2, p_ee_w2);
  pack_b_kernel<<<32, 32, 0, stream>>>(gate_w1, p_gate_w1);
  for (int l = 0; l < 4; ++l) {
    pack_b_kernel<<<96, 32, 0, stream>>>(msg_w1 + (size_t)l * 384 * H, p_msg_w1 + (size_t)l * T384);
    pack_b_kernel<<<32, 32, 0, stream>>>(msg_w2 + (size_t)l * H * H, p_msg_w2 + (size_t)l * T128);
    pack_b_kernel<<<64, 32, 0, stream>>>(upd_w1 + (size_t)l * 2 * H * H, p_upd_w1 + (size_t)l * T256);
    pack_b_kernel<<<32, 32, 0, stream>>>(upd_w2 + (size_t)l * H * H, p_upd_w2 + (size_t)l * T128);
  }

  const int nb_n = (N_ + TM - 1) / TM;
  const int nb_e = (E_ + TM - 1) / TM;
  const int nb_n64 = (N_ + TMB - 1) / TMB;
  const int nb_e64 = (E_ + TMB - 1) / TMB;
  nodeenc_kernel<<<nb_n, 128, 0, stream>>>(node_feat, ne_w1, ne_b1, p_ne_w2, ne_b2, h, N_);
  edgeenc_kernel<<<nb_e, 128, 0, stream>>>(edge_feat, ee_w1, ee_b1, p_ee_w2, ee_b2, e, E_);

  for (int l = 0; l < 4; ++l) {
    hipMemsetAsync(hagg, 0, (size_t)N_ * H * 4, stream);
    msg_kernel<<<nb_e64, 128, 0, stream>>>(h, e, src, dst,
                                           p_msg_w1 + (size_t)l * T384, msg_b1 + l * H,
                                           p_msg_w2 + (size_t)l * T128, msg_b2 + l * H,
                                           hagg, E_);
    upd_kernel<<<nb_n64, 128, 0, stream>>>(h, hagg,
                                           p_upd_w1 + (size_t)l * T256, upd_b1 + l * H,
                                           p_upd_w2 + (size_t)l * T128, upd_b2 + l * H, N_);
  }

  hipMemsetAsync(gmax, 0, B_ * 4, stream);  // 0 == ordered encoding below any real gate
  hipMemsetAsync(z, 0, B_ * 4, stream);
  hipMemsetAsync(emb, 0, (size_t)B_ * H * 4, stream);
  gate_kernel<<<nb_n, 128, 0, stream>>>(h, p_gate_w1, gate_b1, gate_w2, gate_b2, gid,
                                        gate, gmax, N_);
  exp_kernel<<<(N_ + 255) / 256, 256, 0, stream>>>(gate, gid, gmax, exv, z, N_);
  emb_kernel<<<(int)(((size_t)N_ * H + 255) / 256), 256, 0, stream>>>(h, exv, z, gid, emb, N_);
  policy_kernel<<<1, 256, 0, stream>>>(emb, pol_w1, pol_b1, pol_w2, pol_b2, pol_w3,
                                       pol_b3, (float*)d_out);
}
